// PsiForesight_6390911336722
// MI455X (gfx1250) — matrix operand order repaired and checked by reading, not compiled, not run on hardware
//
#include <hip/hip_runtime.h>

// ---------------------------------------------------------------------------
// PsiForesight forward for MI455X (gfx1250).
// Heavy GEMMs: v_wmma_f32_16x16x32_f16 (f16 operands, f32 accum), LDS
// double-buffered, A-tile staged with GLOBAL_LOAD_ASYNC_TO_LDS (ASYNCcnt-
// tracked CDNA5 path). Small ops (LN, fourier-edge MLP, per-channel 8x8 head
// softmax, gate reduction) are fp32 VALU kernels.
// ---------------------------------------------------------------------------

#define Bq  4
#define Tq  16
#define Aq  512
#define Zq  512
#define Eq  64
#define Hq  8
#define HDq 64
static const int Mrows = Bq * Tq * Aq;   // 32768 token rows

typedef __attribute__((ext_vector_type(16))) _Float16 v16h;
typedef __attribute__((ext_vector_type(8)))  _Float16 v8h;
typedef __attribute__((ext_vector_type(8)))  float    v8f;
typedef int b128_t __attribute__((vector_size(16)));   // async-copy payload type

#if defined(__gfx1250__) && __has_builtin(__builtin_amdgcn_global_load_async_to_lds_b128)
#define USE_ASYNC_LDS 1
typedef __attribute__((address_space(1))) b128_t* gas_b128_p;  // global (AS1)
typedef __attribute__((address_space(3))) b128_t* lds_b128_p;  // LDS (AS3)
#else
#define USE_ASYNC_LDS 0
#endif

__device__ __forceinline__ void wait_async0() {
#if USE_ASYNC_LDS
#if __has_builtin(__builtin_amdgcn_s_wait_asynccnt)
  __builtin_amdgcn_s_wait_asynccnt(0);
#else
  asm volatile("s_wait_asynccnt 0" ::: "memory");
#endif
#endif
}

__device__ __forceinline__ float gelu_exact(float x) {
  return 0.5f * x * (1.0f + erff(x * 0.70710678118654752f));
}

// ---------------------------------------------------------------------------
// fp32 -> fp16 conversion (grid-stride)
// ---------------------------------------------------------------------------
__global__ void k_cvt(const float* __restrict__ s, _Float16* __restrict__ d, int n) {
  int i = blockIdx.x * blockDim.x + threadIdx.x;
  int stride = gridDim.x * blockDim.x;
  for (; i < n; i += stride) d[i] = (_Float16)s[i];
}

// ---------------------------------------------------------------------------
// LayerNorm over rows of 512, write f16 (pre-Q projection)
// ---------------------------------------------------------------------------
__global__ __launch_bounds__(128) void k_ln512(const float* __restrict__ x,
                                               const float* __restrict__ g,
                                               const float* __restrict__ b,
                                               _Float16* __restrict__ y) {
  __shared__ float red[128];
  int row = blockIdx.x;
  int t = threadIdx.x;
  const float* xr = x + (size_t)row * Zq;
  float v[4];
#pragma unroll
  for (int j = 0; j < 4; j++) v[j] = xr[t + 128 * j];
  red[t] = v[0] + v[1] + v[2] + v[3];
  __syncthreads();
  for (int o = 64; o > 0; o >>= 1) { if (t < o) red[t] += red[t + o]; __syncthreads(); }
  float mean = red[0] * (1.0f / 512.0f);
  __syncthreads();
  float sq = 0.f;
#pragma unroll
  for (int j = 0; j < 4; j++) { float d0 = v[j] - mean; sq += d0 * d0; }
  red[t] = sq;
  __syncthreads();
  for (int o = 64; o > 0; o >>= 1) { if (t < o) red[t] += red[t + o]; __syncthreads(); }
  float rstd = rsqrtf(red[0] * (1.0f / 512.0f) + 1e-5f);
  _Float16* yr = y + (size_t)row * Zq;
#pragma unroll
  for (int j = 0; j < 4; j++) {
    int c = t + 128 * j;
    yr[c] = (_Float16)((v[j] - mean) * rstd * g[c] + b[c]);
  }
}

// ---------------------------------------------------------------------------
// Edge embedding: fourier(actions) @ W_edge + b_edge, then @ We  -> e[B*T,512]
// ---------------------------------------------------------------------------
__global__ __launch_bounds__(64) void k_edge(const float* __restrict__ actions,
                                             const float* __restrict__ W_edge,
                                             const float* __restrict__ b_edge,
                                             const float* __restrict__ We,
                                             float* __restrict__ eout) {
  __shared__ float feats[38];
  __shared__ float tmp[64];
  int bt = blockIdx.x;
  int t = threadIdx.x;
  if (t == 0) {
    const float* a3 = actions + (size_t)bt * 4;
    float dx = a3[0], dy = a3[1], th = a3[2];
    feats[0] = sinf(th); feats[1] = cosf(th);
    float f = 3.14159265358979323846f;
    for (int k = 0; k < 6; k++) {
      feats[2 + 6 * k + 0] = sinf(f * dx); feats[2 + 6 * k + 1] = cosf(f * dx);
      feats[2 + 6 * k + 2] = sinf(f * dy); feats[2 + 6 * k + 3] = cosf(f * dy);
      feats[2 + 6 * k + 4] = sinf(f * th); feats[2 + 6 * k + 5] = cosf(f * th);
      f *= 2.0f;
    }
  }
  __syncthreads();
  float acc = b_edge[t];
#pragma unroll
  for (int f = 0; f < 38; f++) acc += feats[f] * W_edge[f * 64 + t];
  tmp[t] = acc;
  __syncthreads();
  float* er = eout + (size_t)bt * Zq;
  for (int j = 0; j < 8; j++) {
    int n = t + 64 * j;
    float s = 0.f;
#pragma unroll
    for (int e = 0; e < 64; e++) s += tmp[e] * We[e * Zq + n];
    er[n] = s;
  }
}

// ---------------------------------------------------------------------------
// Per-channel 8-head outer-product softmax mixing:
// out_h[b,t,a,i,d] = sum_j softmax_j(q_i*k_j/8) * v_j   (per d)
// block = 256 threads = 4 rows x 64 channels
// ---------------------------------------------------------------------------
__global__ __launch_bounds__(256) void k_attn(const _Float16* __restrict__ q16,
                                              const _Float16* __restrict__ kc16,
                                              const _Float16* __restrict__ vc16,
                                              const float* __restrict__ e32,
                                              _Float16* __restrict__ outh) {
  int idx = blockIdx.x * 4 + (threadIdx.x >> 6);   // row in [0, B*T*A)
  int d = threadIdx.x & 63;
  int a = idx & (Aq - 1);
  int bt = idx / Aq;            // b*T + t
  int b = bt / Tq;
  const _Float16* qr = q16 + (size_t)idx * Zq;
  const _Float16* kr = kc16 + (size_t)(b * Aq + a) * Zq;
  const _Float16* vr = vc16 + (size_t)(b * Aq + a) * Zq;
  const float* er = e32 + (size_t)bt * Zq;
  float qv[8], kv[8], vv[8];
#pragma unroll
  for (int j = 0; j < 8; j++) {
    int c = j * HDq + d;
    qv[j] = (float)qr[c];
    kv[j] = (float)kr[c] + er[c];
    vv[j] = (float)vr[c];
  }
  _Float16* orow = outh + (size_t)idx * Zq;
#pragma unroll
  for (int i = 0; i < 8; i++) {
    float l[8], mx = -1e30f;
#pragma unroll
    for (int j = 0; j < 8; j++) { l[j] = qv[i] * kv[j] * 0.125f; mx = fmaxf(mx, l[j]); }
    float se = 0.f;
#pragma unroll
    for (int j = 0; j < 8; j++) { l[j] = expf(l[j] - mx); se += l[j]; }
    float o = 0.f;
#pragma unroll
    for (int j = 0; j < 8; j++) o += l[j] * vv[j];
    orow[i * HDq + d] = (_Float16)(o / se);
  }
}

// ---------------------------------------------------------------------------
// Gate second stage: g = sigmoid(dot(G1_row, Wg2) + bg2)
// ---------------------------------------------------------------------------
__global__ __launch_bounds__(128) void k_gate2(const _Float16* __restrict__ g1,
                                               const float* __restrict__ wg2,
                                               const float* __restrict__ bg2,
                                               float* __restrict__ gv) {
  __shared__ float red[128];
  int row = blockIdx.x, t = threadIdx.x;
  const _Float16* xr = g1 + (size_t)row * Zq;
  float s = 0.f;
  for (int j = t; j < Zq; j += 128) s += (float)xr[j] * wg2[j];
  red[t] = s;
  __syncthreads();
  for (int o = 64; o > 0; o >>= 1) { if (t < o) red[t] += red[t + o]; __syncthreads(); }
  if (t == 0) gv[row] = 1.0f / (1.0f + expf(-(red[0] + bg2[0])));
}

// ---------------------------------------------------------------------------
// z_mid = LN(z_pred + g*out), write f32 + f16
// ---------------------------------------------------------------------------
__global__ __launch_bounds__(128) void k_zmid(const float* __restrict__ zp,
                                              const float* __restrict__ out,
                                              const float* __restrict__ gv,
                                              const float* __restrict__ g,
                                              const float* __restrict__ b,
                                              float* __restrict__ y32,
                                              _Float16* __restrict__ y16) {
  __shared__ float red[128];
  int row = blockIdx.x, t = threadIdx.x;
  const float* zr = zp + (size_t)row * Zq;
  const float* orr = out + (size_t)row * Zq;
  float gg = gv[row];
  float v[4];
#pragma unroll
  for (int j = 0; j < 4; j++) v[j] = zr[t + 128 * j] + gg * orr[t + 128 * j];
  red[t] = v[0] + v[1] + v[2] + v[3];
  __syncthreads();
  for (int o = 64; o > 0; o >>= 1) { if (t < o) red[t] += red[t + o]; __syncthreads(); }
  float mean = red[0] * (1.0f / 512.0f);
  __syncthreads();
  float sq = 0.f;
#pragma unroll
  for (int j = 0; j < 4; j++) { float d0 = v[j] - mean; sq += d0 * d0; }
  red[t] = sq;
  __syncthreads();
  for (int o = 64; o > 0; o >>= 1) { if (t < o) red[t] += red[t + o]; __syncthreads(); }
  float rstd = rsqrtf(red[0] * (1.0f / 512.0f) + 1e-5f);
#pragma unroll
  for (int j = 0; j < 4; j++) {
    int c = t + 128 * j;
    float yy = (v[j] - mean) * rstd * g[c] + b[c];
    y32[(size_t)row * Zq + c] = yy;
    y16[(size_t)row * Zq + c] = (_Float16)yy;
  }
}

// ---------------------------------------------------------------------------
// WMMA GEMM: C[M,N] = A[M,K] * B[K,N]   (A,B f16 row-major, fp32 accumulate)
// Block tile 64x128, 8 waves, wave tile 32x32 (2x2 x 16x16x32 frags).
// Double-buffered LDS pipeline: stage tile kt+1 (A async-to-LDS, B via
// register transpose) while WMMAs consume tile kt; s_wait_asynccnt 0 before
// the workgroup barrier. Rolling global pointers (no per-iter 64-bit muls).
// Epilogue: + Cin, + bias[n], gelu, + resid, store f32 and/or f16.
// Requires M%64==0, N%128==0, K%32==0 (true for all call sites here).
// ---------------------------------------------------------------------------
#define BM 64
#define BN 128
#define BK 32
#define LDA 40   // padded LDS row stride (halves): 80B, 16B-aligned chunks
#define LDB 40
#define ATILE (BM * LDA)
#define BTILE (BN * LDB)

__global__ __launch_bounds__(256) void k_gemm(
    const _Float16* __restrict__ A, const _Float16* __restrict__ Bw,
    int M, int N, int K,
    const float* __restrict__ Cin, const float* __restrict__ bias,
    const float* __restrict__ resid, float* __restrict__ out32,
    _Float16* __restrict__ out16, int do_gelu) {
  __shared__ __align__(16) _Float16 sA[2 * ATILE];
  __shared__ __align__(16) _Float16 sB[2 * BTILE];
  (void)M;
  int tid = threadIdx.x;
  int nb = blockIdx.x * BN;
  int mb = blockIdx.y * BM;
  int wave = tid >> 5, lane = tid & 31;
  int wr = wave >> 2, wc = wave & 3;       // 2x4 wave grid
  int l16 = lane & 15;
  bool hi = lane >= 16;

  v8f acc[2][2] = {};

  // cooperative staging assignments
  int arow = tid >> 2;           // 0..63
  int ach  = (tid & 3) * 8;      // 0,8,16,24 (halves)
  int bk0  = tid >> 4;           // 0..15 (pass p adds 16)
  int bc   = (tid & 15) * 8;     // 0..120 step 8

  // rolling global pointers (strength-reduced; +BK / +BK*N per K step)
  const _Float16* aCur = A + (size_t)(mb + arow) * K + ach;
  const _Float16* bCur = Bw + (size_t)bk0 * N + nb + bc;
  const size_t bStep = (size_t)BK * N;

  _Float16* sAw = &sA[arow * LDA + ach];

  // A-fragment LDS offsets (ISA 16-bit A 16x32 layout):
  //   lanes 0-15 : M=lane,   K in {0..7, 16..23}
  //   lanes 16-31: M=lane-16, K in {8..15, 24..31}
  const _Float16* aFrag = &sA[(wr * 32 + l16) * LDA + (hi ? 8 : 0)];
  // B-fragment (K x N, stored transposed in LDS as [n][k]):
  //   lanes 0-15 : N=lane,   K in {0..7, 16..23} ; lanes 16-31: N=lane-16, K in {8..15, 24..31}
  const _Float16* bFrag = &sB[(wc * 32 + l16) * LDB + (hi ? 8 : 0)];

  auto stageA = [&](int buf, const _Float16* src) {
#if USE_ASYNC_LDS
    __builtin_amdgcn_global_load_async_to_lds_b128(
        (gas_b128_p)(void*)src,
        (lds_b128_p)(sAw + buf * ATILE), 0, 0);
#else
    *(v8h*)(sAw + buf * ATILE) = *(const v8h*)src;
#endif
  };
  auto stageB = [&](int buf, const _Float16* src) {
#pragma unroll
    for (int p = 0; p < 2; p++) {
      v8h bv = *(const v8h*)(src + (size_t)p * 16 * N);
      int kk = bk0 + p * 16;
#pragma unroll
      for (int h = 0; h < 8; h++) sB[buf * BTILE + (bc + h) * LDB + kk] = bv[h];
    }
  };

  // prologue: stage tile 0 into buffer 0
  stageA(0, aCur);
  stageB(0, bCur);
  wait_async0();
  __syncthreads();

  const int kIters = K / BK;
  for (int kt = 0; kt < kIters; ++kt) {
    int cur = kt & 1, nxt = cur ^ 1;
    const _Float16* aNxt = aCur + BK;
    const _Float16* bNxt = bCur + bStep;
    if (kt + 1 < kIters) {      // stage next tile while computing this one
      stageA(nxt, aNxt);
      stageB(nxt, bNxt);
      __builtin_prefetch(aNxt + BK, 0, 0);          // global_prefetch_b8, dist 2
      __builtin_prefetch(bNxt + bStep, 0, 0);
    }

    union AF { v16h v; v8h h[2]; };
    AF af[2], bf[2];
#pragma unroll
    for (int i = 0; i < 2; i++) {
      const _Float16* p = aFrag + cur * ATILE + i * 16 * LDA;
      af[i].h[0] = *(const v8h*)p;
      af[i].h[1] = *(const v8h*)(p + 16);
    }
#pragma unroll
    for (int j = 0; j < 2; j++) {
      const _Float16* p = bFrag + cur * BTILE + j * 16 * LDB;
      bf[j].h[0] = *(const v8h*)p;
      bf[j].h[1] = *(const v8h*)(p + 16);
    }
#pragma unroll
    for (int i = 0; i < 2; i++)
#pragma unroll
      for (int j = 0; j < 2; j++)
        acc[i][j] = __builtin_amdgcn_wmma_f32_16x16x32_f16(
            false, af[i].v, false, bf[j].v, (short)0, acc[i][j], false, false);

    wait_async0();      // async A copy for nxt complete before barrier
    __syncthreads();    // (compiler adds dscnt wait for B stores)
    aCur = aNxt;
    bCur = bNxt;
  }

  // Epilogue. C/D layout: lanes 0-15 -> M rows m0+0..7, N=n0+lane;
  //                       lanes 16-31 -> M rows m0+8..15, N=n0+lane-16.
#pragma unroll
  for (int i = 0; i < 2; i++) {
    int mfr = mb + wr * 32 + i * 16 + (hi ? 8 : 0);
#pragma unroll
    for (int j = 0; j < 2; j++) {
      int n = nb + wc * 32 + j * 16 + l16;
#pragma unroll
      for (int r = 0; r < 8; r++) {
        size_t idx = (size_t)(mfr + r) * N + n;
        float x = acc[i][j][r];
        if (Cin)  x += Cin[idx];
        if (bias) x += bias[n];
        if (do_gelu) x = gelu_exact(x);
        if (resid) x += resid[idx];
        if (out32) out32[idx] = x;
        if (out16) out16[idx] = (_Float16)x;
      }
    }
  }
}

// ---------------------------------------------------------------------------
extern "C" void kernel_launch(void* const* d_in, const int* in_sizes, int n_in,
                              void* d_out, int out_size, void* d_ws, size_t ws_size,
                              hipStream_t stream) {
  (void)in_sizes; (void)n_in; (void)out_size; (void)ws_size;
  const float* z_current = (const float*)d_in[0];
  const float* z_pred    = (const float*)d_in[1];
  const float* actions   = (const float*)d_in[2];
  const float* Wq        = (const float*)d_in[3];
  const float* Wk        = (const float*)d_in[4];
  const float* Wv        = (const float*)d_in[5];
  const float* We        = (const float*)d_in[6];
  const float* Wo        = (const float*)d_in[7];
  const float* g_lnq     = (const float*)d_in[8];
  const float* b_lnq     = (const float*)d_in[9];
  const float* g_lno     = (const float*)d_in[10];
  const float* b_lno     = (const float*)d_in[11];
  const float* W_ff1     = (const float*)d_in[12];
  const float* b_ff1     = (const float*)d_in[13];
  const float* W_ff2     = (const float*)d_in[14];
  const float* b_ff2     = (const float*)d_in[15];
  const float* W_edge    = (const float*)d_in[16];
  const float* b_edge    = (const float*)d_in[17];
  const float* Wg1       = (const float*)d_in[18];
  const float* bg1       = (const float*)d_in[19];
  const float* Wg2       = (const float*)d_in[20];
  const float* bg2       = (const float*)d_in[21];

  const int M = Mrows;                 // 32768
  // ---- workspace layout (regions reused along the dataflow) ----
  size_t off = 0;
  char* ws = (char*)d_ws;
  auto take = [&](size_t bytes) -> void* {
    void* p = ws + off; off = (off + bytes + 255) & ~(size_t)255; return p;
  };
  _Float16* wq16   = (_Float16*)take((size_t)512 * 512 * 2);
  _Float16* wk16   = (_Float16*)take((size_t)512 * 512 * 2);
  _Float16* wv16   = (_Float16*)take((size_t)512 * 512 * 2);
  _Float16* wo16   = (_Float16*)take((size_t)512 * 512 * 2);
  _Float16* wg116  = (_Float16*)take((size_t)1024 * 512 * 2);
  _Float16* wff116 = (_Float16*)take((size_t)512 * 2048 * 2);
  _Float16* wff216 = (_Float16*)take((size_t)2048 * 512 * 2);
  _Float16* zpred16 = (_Float16*)take((size_t)M * 512 * 2);  // later: zmid f16
  _Float16* r1      = (_Float16*)take((size_t)M * 512 * 2);  // hln / out_h / G1
  _Float16* zc16    = (_Float16*)take((size_t)2048 * 512 * 2);
  _Float16* kc16    = (_Float16*)take((size_t)2048 * 512 * 2);
  _Float16* vc16    = (_Float16*)take((size_t)2048 * 512 * 2);
  float*    e32     = (float*)take((size_t)64 * 512 * 4);
  _Float16* r2      = (_Float16*)take((size_t)M * 512 * 2);  // q16 / out16
  float*    out32   = (float*)take((size_t)M * 512 * 4);
  float*    r3      = (float*)take((size_t)M * 512 * 4);     // gateC / zmid f32
  float*    gvec    = (float*)take((size_t)M * 4);
  _Float16* ff116   = (_Float16*)take((size_t)M * 2048 * 2);

  _Float16* hln16  = r1;       // LN(z_pred) f16
  _Float16* outh16 = r1;       // attention output (after hln dead)
  _Float16* g116   = r1;       // gelu(gate) f16 (after outh dead)
  _Float16* q16    = r2;
  _Float16* out16  = r2;       // out f16 (after q dead)
  float*    gateC  = r3;
  float*    zmid32 = r3;       // (after gateC dead)
  _Float16* zmid16 = zpred16;  // (after zpred16 dead)

  auto gemm = [&](const _Float16* A_, const _Float16* B_, int M_, int N_, int K_,
                  const float* Cin, const float* bias, const float* resid,
                  float* o32, _Float16* o16, int gel) {
    dim3 grid(N_ / BN, M_ / BM);
    k_gemm<<<grid, 256, 0, stream>>>(A_, B_, M_, N_, K_, Cin, bias, resid, o32, o16, gel);
  };
  auto cvt = [&](const float* s, _Float16* d, int n) {
    int blocks = (n + 255) / 256; if (blocks > 1024) blocks = 1024;
    k_cvt<<<blocks, 256, 0, stream>>>(s, d, n);
  };

  // 1) precision conversion of weights + activations
  cvt(Wq, wq16, 512 * 512);
  cvt(Wk, wk16, 512 * 512);
  cvt(Wv, wv16, 512 * 512);
  cvt(Wo, wo16, 512 * 512);
  cvt(Wg1, wg116, 1024 * 512);
  cvt(W_ff1, wff116, 512 * 2048);
  cvt(W_ff2, wff216, 2048 * 512);
  cvt(z_pred, zpred16, M * 512);
  cvt(z_current, zc16, Bq * Aq * Zq);

  // 2) LN(z_pred) -> f16 (input to Q projection)
  k_ln512<<<M, 128, 0, stream>>>(z_pred, g_lnq, b_lnq, hln16);

  // 3) edge embedding e[b,t,:]
  k_edge<<<Bq * Tq, 64, 0, stream>>>(actions, W_edge, b_edge, We, e32);

  // 4) projections (WMMA GEMMs)
  gemm(zc16, wk16, Bq * Aq, 512, 512, nullptr, nullptr, nullptr, nullptr, kc16, 0);
  gemm(zc16, wv16, Bq * Aq, 512, 512, nullptr, nullptr, nullptr, nullptr, vc16, 0);
  gemm(hln16, wq16, M, 512, 512, nullptr, nullptr, nullptr, nullptr, q16, 0);

  // 5) per-channel 8x8 head softmax mixing
  k_attn<<<M / 4, 256, 0, stream>>>(q16, kc16, vc16, e32, outh16);

  // 6) out = out_h @ Wo  (f32 for residual, f16 for gate GEMM)
  gemm(outh16, wo16, M, 512, 512, nullptr, nullptr, nullptr, out32, out16, 0);

  // 7) gate: concat([out, z_pred]) @ Wg1 as two accumulated K=512 passes
  gemm(out16, wg116, M, 512, 512, nullptr, nullptr, nullptr, gateC, nullptr, 0);
  gemm(zpred16, wg116 + (size_t)512 * 512, M, 512, 512,
       gateC, bg1, nullptr, nullptr, g116, /*gelu=*/1);
  k_gate2<<<M, 128, 0, stream>>>(g116, Wg2, bg2, gvec);

  // 8) z_mid = LN(z_pred + g*out)
  k_zmid<<<M, 128, 0, stream>>>(z_pred, out32, gvec, g_lno, b_lno, zmid32, zmid16);

  // 9) FFN: gelu(z_mid@W_ff1 + b_ff1) @ W_ff2 + b_ff2 + z_mid -> d_out (f32)
  gemm(zmid16, wff116, M, 2048, 512, nullptr, b_ff1, nullptr, nullptr, ff116, /*gelu=*/1);
  gemm(ff116, wff216, M, 512, 2048, nullptr, b_ff2, zmid32, (float*)d_out, nullptr, 0);
}
